// MultiModalPerceiverResamplerBlock_27453430956141
// MI455X (gfx1250) — matrix operand order repaired and checked by reading, not compiled, not run on hardware
//
#include <hip/hip_runtime.h>
#include <hip/hip_bf16.h>
#include <math.h>

// MI455X / gfx1250: wave32, WMMA bf16 16x16x32 fp32-acc, async global->LDS copies.

typedef __bf16 bf16;
typedef __attribute__((ext_vector_type(16))) __bf16 v16bf;
typedef __attribute__((ext_vector_type(8)))  float  v8f;
typedef __attribute__((ext_vector_type(4)))  unsigned int u32x4;

union FragBF { u32x4 u[2]; v16bf v; };

#define TPB 256

static __device__ __forceinline__ unsigned lds_addr32(const void* p) {
  // generic -> LDS(AS3) address (byte offset within workgroup LDS)
  return (unsigned)(unsigned long long)(__attribute__((address_space(3))) const void*)p;
}

static __device__ __forceinline__ void async_b128(unsigned lds_off,
                                                  const void* sbase,
                                                  unsigned vbyte_off) {
  // per-lane 16B global->LDS async copy, tracked by ASYNCcnt
  asm volatile("global_load_async_to_lds_b128 %0, %1, %2"
               :: "v"(lds_off), "v"(vbyte_off), "s"(sbase) : "memory");
}

// ------------------------------------------------------------------
// fp32 -> bf16 convert (row-major copy)
// ------------------------------------------------------------------
__global__ __launch_bounds__(TPB) void cvt_kernel(const float* __restrict__ in,
                                                  bf16* __restrict__ out, size_t n) {
  size_t i  = (size_t)blockIdx.x * TPB + threadIdx.x;
  size_t st = (size_t)gridDim.x * TPB;
  for (; i < n; i += st) out[i] = (bf16)in[i];
}

// ------------------------------------------------------------------
// fp32 RxC -> bf16 CxR transpose (32x32 tiles)
// ------------------------------------------------------------------
__global__ __launch_bounds__(TPB) void tcvt_kernel(const float* __restrict__ in,
    bf16* __restrict__ out, int R, int C) {
  __shared__ float t[32][33];
  int bx = blockIdx.x * 32;   // input col tile
  int by = blockIdx.y * 32;   // input row tile
  for (int i = threadIdx.x; i < 32*32; i += TPB) {
    int r = i >> 5, c = i & 31;
    t[r][c] = in[(size_t)(by + r)*C + (bx + c)];
  }
  __syncthreads();
  for (int i = threadIdx.x; i < 32*32; i += TPB) {
    int r = i >> 5, c = i & 31;
    out[(size_t)(bx + r)*R + (by + c)] = (bf16)t[c][r];
  }
}

// ------------------------------------------------------------------
// LayerNorm (row per block), fp32 in -> bf16 out
// ------------------------------------------------------------------
__global__ __launch_bounds__(TPB) void ln_kernel(const float* __restrict__ x,
    const float* __restrict__ sc, const float* __restrict__ of,
    bf16* __restrict__ out, int D) {
  __shared__ float red[TPB];
  int row = blockIdx.x;
  const float* xr = x + (size_t)row * D;
  float s = 0.f;
  for (int i = threadIdx.x; i < D; i += TPB) s += xr[i];
  red[threadIdx.x] = s; __syncthreads();
  for (int st = TPB/2; st > 0; st >>= 1) {
    if ((int)threadIdx.x < st) red[threadIdx.x] += red[threadIdx.x+st];
    __syncthreads();
  }
  float mu = red[0] / D;
  __syncthreads();
  float s2 = 0.f;
  for (int i = threadIdx.x; i < D; i += TPB) { float d = xr[i]-mu; s2 += d*d; }
  red[threadIdx.x] = s2; __syncthreads();
  for (int st = TPB/2; st > 0; st >>= 1) {
    if ((int)threadIdx.x < st) red[threadIdx.x] += red[threadIdx.x+st];
    __syncthreads();
  }
  float rstd = rsqrtf(red[0]/D + 1e-5f);
  bf16* orow = out + (size_t)row * D;
  for (int i = threadIdx.x; i < D; i += TPB)
    orow[i] = (bf16)((xr[i]-mu)*rstd*sc[i] + of[i]);
}

// ------------------------------------------------------------------
// Softmax over T real logits + 1 extra learned-bias logit.
// Row per block; bf16 weights written in-place (row-local, safe).
// ------------------------------------------------------------------
__global__ __launch_bounds__(TPB) void softmax_kernel(
    float* __restrict__ logits, const bf16* __restrict__ qh,
    const float* __restrict__ bias_k, float* __restrict__ wextra,
    float alpha, int T, int Q, int H, int HK, int K) {
  __shared__ float red[TPB];
  int rowid = blockIdx.x;
  int z = rowid / Q, q = rowid % Q;
  int b = z / H,     h = z % H;
  float* L = logits + (size_t)rowid * T;

  float pe = 0.f;
  for (int i = threadIdx.x; i < K; i += TPB)
    pe += (float)qh[((size_t)(b*Q + q))*HK + h*K + i] * bias_k[h*K + i];
  red[threadIdx.x] = pe; __syncthreads();
  for (int st = TPB/2; st > 0; st >>= 1) {
    if ((int)threadIdx.x < st) red[threadIdx.x] += red[threadIdx.x+st];
    __syncthreads();
  }
  float e = red[0] * alpha;
  __syncthreads();

  float mx = e;
  for (int i = threadIdx.x; i < T; i += TPB) mx = fmaxf(mx, L[i]);
  red[threadIdx.x] = mx; __syncthreads();
  for (int st = TPB/2; st > 0; st >>= 1) {
    if ((int)threadIdx.x < st) red[threadIdx.x] = fmaxf(red[threadIdx.x], red[threadIdx.x+st]);
    __syncthreads();
  }
  mx = red[0];
  __syncthreads();

  float pbuf[8];  // T/TPB == 8
  float sum = 0.f; int c = 0;
  for (int i = threadIdx.x; i < T; i += TPB) {
    float pv = expf(L[i]-mx);
    pbuf[c++] = pv; sum += pv;
  }
  red[threadIdx.x] = sum; __syncthreads();
  for (int st = TPB/2; st > 0; st >>= 1) {
    if ((int)threadIdx.x < st) red[threadIdx.x] += red[threadIdx.x+st];
    __syncthreads();
  }
  float ee  = expf(e - mx);
  float inv = 1.f / (red[0] + ee);

  bf16* W = (bf16*)L;
  c = 0;
  for (int i = threadIdx.x; i < T; i += TPB) W[i] = (bf16)(pbuf[c++]*inv);
  if (threadIdx.x == 0) wextra[rowid] = ee * inv;
}

// ------------------------------------------------------------------
// WMMA GEMM, all-BT operands: C = alpha*(A @ Bt^T) [+bias][+epi]
//   A : M x Kd bf16 row-major (lda)
//   Bt: N x Kd bf16 row-major (ldb)  (K-contiguous -> async-copy friendly)
// Block 128x128, K-step 32, double-buffered LDS via
// global_load_async_to_lds_b128 (ASYNCcnt), branch-free steady-state
// loop with peeled tail. 8 waves in 2(M)x4(N), each wave 64x32.
//   EPI: 0 none | 1 +res[row*ldc+col] | 2 GELU | 3 +exw[z*M+row]*bvv[h*N+col]
//   OUTT: write C transposed: C[col*ldc + row]
// ------------------------------------------------------------------
#define BMT 128
#define BNT 128
#define BKT 32
#define LDSROW 40            // padded row (bf16 elems); 80B, 16B-aligned
#define BUFSZ  (128*LDSROW)  // elems per buffer

template<bool BIAS, int EPI, bool OUTBF, bool OUTT>
__global__ __launch_bounds__(TPB) void gemm_wmma(
    const bf16* __restrict__ A, const bf16* __restrict__ Bt, void* __restrict__ Cv,
    const float* __restrict__ bias, const float* __restrict__ res,
    const float* __restrict__ exw,  const float* __restrict__ bvv,
    int M, int N, int Kd, int lda, int ldb, int ldc,
    size_t sAb, size_t sAh, size_t sBb, size_t sBh, size_t sCb, size_t sCh,
    int Hdiv, float alpha) {
  int z  = blockIdx.z;
  int bi = z / Hdiv, hh = z % Hdiv;
  A  += (size_t)bi*sAb + (size_t)hh*sAh;
  Bt += (size_t)bi*sBb + (size_t)hh*sBh;
  size_t cOff = (size_t)bi*sCb + (size_t)hh*sCh;

  __shared__ __align__(16) bf16 As[2][128][LDSROW];
  __shared__ __align__(16) bf16 Bs[2][128][LDSROW];

  int tid  = threadIdx.x;
  int wave = tid >> 5;
  int lane = tid & 31;
  int lh   = lane >> 4;
  int ll   = lane & 15;
  int wM   = wave & 1;        // 2 waves along M
  int wN   = wave >> 1;       // 4 waves along N

  int m0 = blockIdx.y * BMT;
  int n0 = blockIdx.x * BNT;

  unsigned baseA = lds_addr32(&As[0][0][0]);
  unsigned baseB = lds_addr32(&Bs[0][0][0]);

  // this thread's two 16B chunks per tile: chunk id = j*256+tid -> (row, kchunk)
  int r0 = tid >> 2,         c0 = tid & 3;
  int r1 = (256 + tid) >> 2, c1 = (256 + tid) & 3;   // = r0+64, c0

  v8f acc[4][2];
  #pragma unroll
  for (int i = 0; i < 4; ++i)
    #pragma unroll
    for (int n = 0; n < 2; ++n)
      #pragma unroll
      for (int j = 0; j < 8; ++j) acc[i][n][j] = 0.f;

  auto issue = [&](int buf, int k0) {
    unsigned bufoff = (unsigned)buf * (BUFSZ*2u);  // bytes
    async_b128(baseA + bufoff + (unsigned)(r0*LDSROW*2 + c0*16), A,
               (unsigned)(((m0 + r0)*lda + k0 + c0*8) * 2));
    async_b128(baseA + bufoff + (unsigned)(r1*LDSROW*2 + c1*16), A,
               (unsigned)(((m0 + r1)*lda + k0 + c1*8) * 2));
    async_b128(baseB + bufoff + (unsigned)(r0*LDSROW*2 + c0*16), Bt,
               (unsigned)(((n0 + r0)*ldb + k0 + c0*8) * 2));
    async_b128(baseB + bufoff + (unsigned)(r1*LDSROW*2 + c1*16), Bt,
               (unsigned)(((n0 + r1)*ldb + k0 + c1*8) * 2));
  };

  auto compute = [&](int cur) {
    FragBF fa[4], fb[2];
    #pragma unroll
    for (int mt = 0; mt < 4; ++mt) {
      int ar = wM*64 + mt*16 + ll;
      fa[mt].u[0] = *(const u32x4*)(&As[cur][ar][lh*8]);
      fa[mt].u[1] = *(const u32x4*)(&As[cur][ar][16 + lh*8]);
    }
    #pragma unroll
    for (int nt = 0; nt < 2; ++nt) {
      int br = wN*32 + nt*16 + ll;
      fb[nt].u[0] = *(const u32x4*)(&Bs[cur][br][lh*8]);
      fb[nt].u[1] = *(const u32x4*)(&Bs[cur][br][16 + lh*8]);
    }
    #pragma unroll
    for (int mt = 0; mt < 4; ++mt)
      #pragma unroll
      for (int nt = 0; nt < 2; ++nt)
        acc[mt][nt] = __builtin_amdgcn_wmma_f32_16x16x32_bf16(
            false, fa[mt].v, false, fb[nt].v, (short)0, acc[mt][nt], false, false);
  };

  issue(0, 0);

  // branch-free steady state: prefetch next tile, retire oldest, compute
  int k0 = 0;
  for (; k0 < Kd - BKT; k0 += BKT) {
    int cur = (k0 >> 5) & 1;
    issue(cur ^ 1, k0 + BKT);
    asm volatile("s_wait_asynccnt 0x4" ::: "memory");  // oldest 4 (cur tile) done
    __syncthreads();
    compute(cur);
    __syncthreads();   // all waves done reading cur before it is overwritten
  }
  // peeled tail
  {
    int cur = (k0 >> 5) & 1;
    asm volatile("s_wait_asynccnt 0x0" ::: "memory");
    __syncthreads();
    compute(cur);
  }

  // Epilogue. D layout: lane ll = N, VGPR j -> M = j + lh*8
  #pragma unroll
  for (int nt = 0; nt < 2; ++nt) {
    int gn = n0 + wN*32 + nt*16 + ll;
    float bvl = BIAS ? bias[gn] : 0.f;
    float r1e = (EPI == 3) ? bvv[hh*N + gn] : 0.f;
    #pragma unroll
    for (int mt = 0; mt < 4; ++mt) {
      #pragma unroll
      for (int j = 0; j < 8; ++j) {
        int gm = m0 + wM*64 + mt*16 + lh*8 + j;
        float v = alpha * acc[mt][nt][j] + bvl;
        if (EPI == 1) v += res[(size_t)gm*ldc + gn];
        if (EPI == 2) v = 0.5f * v * (1.0f + erff(v * 0.70710678118654752f));
        if (EPI == 3) v += exw[(size_t)z*M + gm] * r1e;
        size_t ci = OUTT ? (cOff + (size_t)gn*ldc + gm)
                         : (cOff + (size_t)gm*ldc + gn);
        if (OUTBF) ((bf16*)Cv)[ci] = (bf16)v;
        else       ((float*)Cv)[ci] = v;
      }
    }
  }
}

// ------------------------------------------------------------------
// Host-side helpers
// ------------------------------------------------------------------
template<bool BIAS, int EPI, bool OUTBF, bool OUTT>
static void launch_gemm(hipStream_t s, const bf16* A, const bf16* Bt, void* C,
    const float* bias, const float* res, const float* exw, const float* bvv,
    int M, int N, int Kd, int lda, int ldb, int ldc,
    size_t sAb, size_t sAh, size_t sBb, size_t sBh, size_t sCb, size_t sCh,
    int Z, int Hdiv, float alpha) {
  dim3 g(N / BNT, M / BMT, Z);
  gemm_wmma<BIAS,EPI,OUTBF,OUTT><<<g, dim3(TPB), 0, s>>>(
      A, Bt, C, bias, res, exw, bvv, M, N, Kd, lda, ldb, ldc,
      sAb, sAh, sBb, sBh, sCb, sCh, Hdiv, alpha);
}

namespace {
constexpr int Bc = 4, Qc = 256, Sc = 2048, Dc = 2048, Hc = 16, Kc = 128, Fc = 8192;
constexpr int HKc = Hc * Kc;           // 2048
constexpr int Tc  = Sc;                // 2048
constexpr float ALPHA = 0.08838834764831845f; // 1/sqrt(128)
}

static void run_attn(hipStream_t s,
    const float* x_in, const bf16* xf_b,
    const float* ln_s, const float* ln_o,
    const bf16* wqT, const float* bq,
    const bf16* wkT, const float* bk,
    const bf16* wvT, const float* bv,
    const bf16* woT, const float* bo,
    const float* bias_k, const float* bias_v,
    bf16* hb, bf16* qhb, bf16* khb, bf16* vhT,
    float* logits, float* wextra, bf16* ctxb,
    float* x_out) {
  ln_kernel<<<Bc*Qc, TPB, 0, s>>>(x_in, ln_s, ln_o, hb, Dc);
  // qh = h @ wq + bq   [1024 x 2048]   (wqT: HK x D)
  launch_gemm<true,0,true,false>(s, hb, wqT, qhb, bq, nullptr, nullptr, nullptr,
      Bc*Qc, HKc, Dc, Dc, Dc, HKc, 0,0,0,0,0,0, 1, 1, 1.f);
  // kh = xf @ wk + bk  [8192 x 2048]
  launch_gemm<true,0,true,false>(s, xf_b, wkT, khb, bk, nullptr, nullptr, nullptr,
      Bc*Sc, HKc, Dc, Dc, Dc, HKc, 0,0,0,0,0,0, 1, 1, 1.f);
  // vhT = (xf @ wv + bv)^T   -> [HK x B*S]
  launch_gemm<true,0,true,true>(s, xf_b, wvT, vhT, bv, nullptr, nullptr, nullptr,
      Bc*Sc, HKc, Dc, Dc, Dc, Bc*Sc, 0,0,0,0,0,0, 1, 1, 1.f);
  // logits[b,h] = qh[b,:,h,:] @ kh[b,:,h,:]^T / sqrt(K)  per (b,h): 256x2048, Kd=128
  launch_gemm<false,0,false,false>(s, qhb, khb, logits, nullptr, nullptr, nullptr, nullptr,
      Qc, Tc, Kc, HKc, HKc, Tc,
      (size_t)Qc*HKc, (size_t)Kc,
      (size_t)Sc*HKc, (size_t)Kc,
      (size_t)Hc*Qc*Tc, (size_t)Qc*Tc,
      Bc*Hc, Hc, ALPHA);
  softmax_kernel<<<Bc*Hc*Qc, TPB, 0, s>>>(logits, qhb, bias_k, wextra,
      ALPHA, Tc, Qc, Hc, HKc, Kc);
  // ctx[b,:,h,:] = weights[b,h] @ vh[b,:,h,:] + wextra*bias_v  per (b,h): 256x128, Kd=2048
  launch_gemm<false,3,true,false>(s, (const bf16*)logits, vhT, ctxb,
      nullptr, nullptr, wextra, bias_v,
      Qc, Kc, Tc, 2*Tc, Bc*Sc, HKc,
      (size_t)Hc*Qc*2*Tc, (size_t)Qc*2*Tc,
      (size_t)Sc, (size_t)Kc*(Bc*Sc),
      (size_t)Qc*HKc, (size_t)Kc,
      Bc*Hc, Hc, 1.f);
  // x_out = x_in + ctx @ wo + bo   [1024 x 2048]  (woT: D x HK)
  launch_gemm<true,1,false,false>(s, ctxb, woT, x_out, bo, x_in, nullptr, nullptr,
      Bc*Qc, Dc, HKc, HKc, HKc, Dc, 0,0,0,0,0,0, 1, 1, 1.f);
}

extern "C" void kernel_launch(void* const* d_in, const int* in_sizes, int n_in,
                              void* d_out, int out_size, void* d_ws, size_t ws_size,
                              hipStream_t stream) {
  (void)in_sizes; (void)n_in; (void)out_size; (void)ws_size;

  const float* x       = (const float*)d_in[0];
  const float* xf1     = (const float*)d_in[1];
  const float* xf2     = (const float*)d_in[2];
  const float* ln1_s   = (const float*)d_in[3];
  const float* ln1_o   = (const float*)d_in[4];
  const float* wq1     = (const float*)d_in[5];
  const float* bq1     = (const float*)d_in[6];
  const float* wk1     = (const float*)d_in[7];
  const float* bk1     = (const float*)d_in[8];
  const float* wv1     = (const float*)d_in[9];
  const float* bv1     = (const float*)d_in[10];
  const float* wo1     = (const float*)d_in[11];
  const float* bo1     = (const float*)d_in[12];
  const float* biask1  = (const float*)d_in[13];
  const float* biasv1  = (const float*)d_in[14];
  const float* ln2_s   = (const float*)d_in[15];
  const float* ln2_o   = (const float*)d_in[16];
  const float* wq2     = (const float*)d_in[17];
  const float* bq2     = (const float*)d_in[18];
  const float* wk2     = (const float*)d_in[19];
  const float* bk2     = (const float*)d_in[20];
  const float* wv2     = (const float*)d_in[21];
  const float* bv2     = (const float*)d_in[22];
  const float* wo2     = (const float*)d_in[23];
  const float* bo2     = (const float*)d_in[24];
  const float* biask2  = (const float*)d_in[25];
  const float* biasv2  = (const float*)d_in[26];
  const float* lnm_s   = (const float*)d_in[27];
  const float* lnm_o   = (const float*)d_in[28];
  const float* wfc1    = (const float*)d_in[29];
  const float* bfc1    = (const float*)d_in[30];
  const float* wfc2    = (const float*)d_in[31];
  const float* bfc2    = (const float*)d_in[32];

  char* p = (char*)d_ws;
  auto carve = [&](size_t bytes) -> void* {
    void* r = (void*)p;
    p += (bytes + 255) & ~(size_t)255;
    return r;
  };
  const size_t WSZ  = (size_t)Dc * HKc;      // 4 Mi
  const size_t FCSZ = (size_t)Dc * Fc;       // 16 Mi
  const size_t XFSZ = (size_t)Bc * Sc * Dc;  // 16 Mi
  const size_t XQSZ = (size_t)Bc * Qc * Dc;  //  2 Mi

  bf16* wq1T  = (bf16*)carve(WSZ*2);  bf16* wk1T = (bf16*)carve(WSZ*2);
  bf16* wv1T  = (bf16*)carve(WSZ*2);  bf16* wo1T = (bf16*)carve(WSZ*2);
  bf16* wq2T  = (bf16*)carve(WSZ*2);  bf16* wk2T = (bf16*)carve(WSZ*2);
  bf16* wv2T  = (bf16*)carve(WSZ*2);  bf16* wo2T = (bf16*)carve(WSZ*2);
  bf16* wfc1T = (bf16*)carve(FCSZ*2); bf16* wfc2T = (bf16*)carve(FCSZ*2);
  bf16* xf1b  = (bf16*)carve(XFSZ*2); bf16* xf2b  = (bf16*)carve(XFSZ*2);
  bf16* hb    = (bf16*)carve(XQSZ*2);
  bf16* qhb   = (bf16*)carve(XQSZ*2);
  bf16* khb   = (bf16*)carve(XFSZ*2);
  bf16* vhT   = (bf16*)carve(XFSZ*2);
  float* logits = (float*)carve((size_t)Bc*Hc*Qc*Tc*4);  // 134 MB; bf16 weights reuse it
  float* wextra = (float*)carve((size_t)Bc*Hc*Qc*4);
  bf16*  ctxb   = (bf16*)carve(XQSZ*2);
  float* x1     = (float*)carve(XQSZ*4);
  float* x2     = (float*)carve(XQSZ*4);
  bf16*  gb     = (bf16*)carve((size_t)Bc*Qc*Fc*2);

  // weights: fp32 RxC -> bf16 CxR (N-major, K-contiguous for BT GEMM)
  auto tcvt = [&](const float* in, bf16* out, int R, int C) {
    dim3 g(C/32, R/32);
    tcvt_kernel<<<g, TPB, 0, stream>>>(in, out, R, C);
  };
  tcvt(wq1, wq1T, Dc, HKc);  tcvt(wk1, wk1T, Dc, HKc);
  tcvt(wv1, wv1T, Dc, HKc);  tcvt(wo1, wo1T, HKc, Dc);
  tcvt(wq2, wq2T, Dc, HKc);  tcvt(wk2, wk2T, Dc, HKc);
  tcvt(wv2, wv2T, Dc, HKc);  tcvt(wo2, wo2T, HKc, Dc);
  tcvt(wfc1, wfc1T, Dc, Fc); tcvt(wfc2, wfc2T, Fc, Dc);

  cvt_kernel<<<4096, TPB, 0, stream>>>(xf1, xf1b, XFSZ);
  cvt_kernel<<<4096, TPB, 0, stream>>>(xf2, xf2b, XFSZ);

  run_attn(stream, x, xf1b, ln1_s, ln1_o, wq1T, bq1, wk1T, bk1, wv1T, bv1,
           wo1T, bo1, biask1, biasv1, hb, qhb, khb, vhT, logits, wextra, ctxb, x1);

  run_attn(stream, x1, xf2b, ln2_s, ln2_o, wq2T, bq2, wk2T, bk2, wv2T, bv2,
           wo2T, bo2, biask2, biasv2, hb, qhb, khb, vhT, logits, wextra, ctxb, x2);

  // MLP
  ln_kernel<<<Bc*Qc, TPB, 0, stream>>>(x2, lnm_s, lnm_o, hb, Dc);
  launch_gemm<true,2,true,false>(stream, hb, wfc1T, gb, bfc1, nullptr, nullptr, nullptr,
      Bc*Qc, Fc, Dc, Dc, Dc, Fc, 0,0,0,0,0,0, 1, 1, 1.f);
  launch_gemm<true,1,false,false>(stream, gb, wfc2T, d_out, bfc2, x2, nullptr, nullptr,
      Bc*Qc, Dc, Fc, Fc, Fc, Dc, 0,0,0,0,0,0, 1, 1, 1.f);
}
